// GlobalInteractor_80942953661182
// MI455X (gfx1250) — compile-verified, matrix operand order checked by reading, not run on hardware
//
#include <hip/hip_runtime.h>
#include <hip/hip_bf16.h>
#include <stdint.h>

// ---------------- constants ----------------
#define BB 8
#define NN 192
#define DD 128
#define HH 8
#define LL 3
#define FFD 512
#define MM (BB*NN)          // 1536 token rows
#define NPAIR (BB*NN*NN)    // 294912 pair rows

typedef __attribute__((ext_vector_type(16))) _Float16 v16h;
typedef __attribute__((ext_vector_type(8)))  float    v8f;
typedef __attribute__((ext_vector_type(4)))  unsigned int v4u;

union V16HU { v16h h; v4u q[2]; };

// Fragment loader matching CDNA5 16-bit A/B layouts:
// lane (m = lane&15, g = lane>>4) holds halves k = kb*32 + 8g + {0..7} and
// kb*32 + 16 + 8g + {0..7}  => two contiguous 16-byte loads from the row.
__device__ __forceinline__ v16h load_frag(const _Float16* rowPtr, int kb) {
    int g = (threadIdx.x >> 4) & 1;
    int k0 = kb * 32 + 8 * g;
    V16HU f;
    f.q[0] = *(const v4u*)(rowPtr + k0);
    f.q[1] = *(const v4u*)(rowPtr + k0 + 16);
    return f.h;
}

__device__ __forceinline__ v8f wmma_f16(v16h a, v16h b, v8f c) {
    // v_wmma_f32_16x16x32_f16  (neg_a, A, neg_b, B, c_mod, C, reuse_a, reuse_b)
    return __builtin_amdgcn_wmma_f32_16x16x32_f16(false, a, false, b, (short)0, c, false, false);
}

// ---------------- weight transpose + f32->f16 ----------------
__global__ void tc_kernel(const float* __restrict__ src, _Float16* __restrict__ dst,
                          int R, int C) {
    int idx = blockIdx.x * blockDim.x + threadIdx.x;
    if (idx < R * C) {
        int r = idx / C, c = idx % C;
        dst[(size_t)c * R + r] = (_Float16)src[idx];   // dst: [C][R]
    }
}

// ---------------- LayerNorm over D=128, one row per wave ----------------
__global__ void __launch_bounds__(256) ln_kernel(const float* __restrict__ x,
                                                 const float* __restrict__ g,
                                                 const float* __restrict__ b,
                                                 float* __restrict__ out, int M) {
    int wave = threadIdx.x >> 5;
    int lane = threadIdx.x & 31;
    int row = blockIdx.x * 8 + wave;
    if (row >= M) return;
    const float* xr = x + (size_t)row * DD;
    float v[4]; float s = 0.f;
    #pragma unroll
    for (int t = 0; t < 4; ++t) { v[t] = xr[lane + 32 * t]; s += v[t]; }
    #pragma unroll
    for (int m = 16; m >= 1; m >>= 1) s += __shfl_xor(s, m, 32);
    float mean = s * (1.f / 128.f);
    float var = 0.f;
    #pragma unroll
    for (int t = 0; t < 4; ++t) { float d = v[t] - mean; var += d * d; }
    #pragma unroll
    for (int m = 16; m >= 1; m >>= 1) var += __shfl_xor(var, m, 32);
    float rs = rsqrtf(var * (1.f / 128.f) + 1e-5f);
    #pragma unroll
    for (int t = 0; t < 4; ++t) {
        int c = lane + 32 * t;
        out[(size_t)row * DD + c] = (v[t] - mean) * rs * g[c] + b[c];
    }
}

// ---------------- generic 16-row WMMA GEMM: C = act(A@Wt^T + bias) (+resid) --
// A: [M][K] f32, Wt: [Nout][K] f16.  grid: (M/16, Nout/128), block 256 (8 waves)
__global__ void __launch_bounds__(256) gemm16_kernel(const float* __restrict__ A,
                                                     const _Float16* __restrict__ Wt,
                                                     const float* __restrict__ bias,
                                                     const float* __restrict__ resid,
                                                     float* __restrict__ C,
                                                     int K, int Nout, int act) {
    __shared__ _Float16 a_s[16 * 520];               // K up to 512, stride K+8
    const int SA = K + 8;
    for (int e = threadIdx.x; e < 16 * K; e += 256) {
        int r = e / K, c = e % K;
        a_s[r * SA + c] = (_Float16)A[((size_t)blockIdx.x * 16 + r) * K + c];
    }
    __syncthreads();

    int w = threadIdx.x >> 5, lane = threadIdx.x & 31;
    int l16 = lane & 15, g = lane >> 4;
    int col = blockIdx.y * 128 + w * 16 + l16;
    const _Float16* arow = a_s + l16 * SA;
    const _Float16* brow = Wt + (size_t)col * K;
    v8f acc = {};
    for (int kb = 0; kb < K / 32; ++kb)
        acc = wmma_f16(load_frag(arow, kb), load_frag(brow, kb), acc);

    float bv = bias ? bias[col] : 0.f;
    #pragma unroll
    for (int rr = 0; rr < 8; ++rr) {
        int row = blockIdx.x * 16 + rr + 8 * g;
        float val = acc[rr] + bv;
        if (resid) val += resid[(size_t)row * Nout + col];
        if (act) val = fmaxf(val, 0.f);
        C[(size_t)row * Nout + col] = val;
    }
}

// ---------------- rel_pe: fused geometry + 3-stage MLP with LN -------------
__device__ __forceinline__ void ln_rows16(const float* hf, _Float16* hh,
                                          const float* gg, const float* bb,
                                          bool relu) {
    // 8 waves x 2 rows; each 16-lane half-wave normalizes one row of 128
    int w = threadIdx.x >> 5, lane = threadIdx.x & 31;
    int g = lane >> 4, l16 = lane & 15;
    int r = 2 * w + g;
    float vals[8]; float s = 0.f;
    #pragma unroll
    for (int cc = 0; cc < 8; ++cc) { vals[cc] = hf[r * 132 + l16 * 8 + cc]; s += vals[cc]; }
    #pragma unroll
    for (int m = 8; m >= 1; m >>= 1) s += __shfl_xor(s, m, 32);
    float mean = s * (1.f / 128.f);
    float var = 0.f;
    #pragma unroll
    for (int cc = 0; cc < 8; ++cc) { float d = vals[cc] - mean; var += d * d; }
    #pragma unroll
    for (int m = 8; m >= 1; m >>= 1) var += __shfl_xor(var, m, 32);
    float rs = rsqrtf(var * (1.f / 128.f) + 1e-5f);
    #pragma unroll
    for (int cc = 0; cc < 8; ++cc) {
        int c = l16 * 8 + cc;
        float y = (vals[cc] - mean) * rs * gg[c] + bb[c];
        if (relu) y = fmaxf(y, 0.f);
        hh[r * 136 + c] = (_Float16)y;
    }
}

__device__ __forceinline__ void mlp_gemm16(const _Float16* hh, const _Float16* wt,
                                           const float* bias, float* hf) {
    int w = threadIdx.x >> 5, lane = threadIdx.x & 31;
    int l16 = lane & 15, g = lane >> 4;
    const _Float16* arow = hh + l16 * 136;
    const _Float16* brow = wt + (size_t)(w * 16 + l16) * 128;
    v8f acc = {};
    #pragma unroll
    for (int kb = 0; kb < 4; ++kb)
        acc = wmma_f16(load_frag(arow, kb), load_frag(brow, kb), acc);
    int col = w * 16 + l16;
    float bv = bias[col];
    #pragma unroll
    for (int rr = 0; rr < 8; ++rr)
        hf[(rr + 8 * g) * 132 + col] = acc[rr] + bv;
}

__global__ void __launch_bounds__(256) relpe_kernel(
    const float* __restrict__ pos, const float* __restrict__ heading,
    const float* __restrict__ w1, const float* __restrict__ b1,
    const float* __restrict__ g1, const float* __restrict__ bb1,
    const _Float16* __restrict__ w2t, const float* __restrict__ b2,
    const float* __restrict__ g2, const float* __restrict__ bb2,
    const _Float16* __restrict__ w3t, const float* __restrict__ b3,
    const float* __restrict__ g3, const float* __restrict__ bb3,
    _Float16* __restrict__ relpe) {
    __shared__ float    hf[16 * 132];
    __shared__ _Float16 hh[16 * 136];
    const int P0 = blockIdx.x * 16;
    const int b  = P0 / (NN * NN);
    const int rem = P0 % (NN * NN);
    const int i  = rem / NN;
    const int j0 = rem % NN;
    {   // stage 1: geo (4) -> 128, scalar FMAs (tiny K)
        int r = threadIdx.x >> 4, part = threadIdx.x & 15;
        int j = j0 + r;
        float cx = pos[(b * NN + i) * 2 + 0], cy = pos[(b * NN + i) * 2 + 1];
        float hd = heading[b * NN + i];
        float dx = pos[(b * NN + j) * 2 + 0] - cx;
        float dy = pos[(b * NN + j) * 2 + 1] - cy;
        float ch = __cosf(hd), sh = __sinf(hd);
        float lx =  dx * ch + dy * sh;
        float ly = -dx * sh + dy * ch;
        float rt = heading[b * NN + j] - hd;
        float c2 = __cosf(rt), s2 = __sinf(rt);
        #pragma unroll
        for (int cc = 0; cc < 8; ++cc) {
            int c = part * 8 + cc;
            hf[r * 132 + c] = lx * w1[c] + ly * w1[128 + c]
                            + c2 * w1[256 + c] + s2 * w1[384 + c] + b1[c];
        }
    }
    __syncthreads();
    ln_rows16(hf, hh, g1, bb1, true);
    __syncthreads();
    mlp_gemm16(hh, w2t, b2, hf);        // stage 2 (WMMA)
    __syncthreads();
    ln_rows16(hf, hh, g2, bb2, true);
    __syncthreads();
    mlp_gemm16(hh, w3t, b3, hf);        // stage 3 (WMMA)
    __syncthreads();
    ln_rows16(hf, hh, g3, bb3, false);
    __syncthreads();
    {   // write f16 rel_pe row-major [pair][128]
        int r = threadIdx.x >> 4, part = threadIdx.x & 15;
        *(v4u*)(relpe + (size_t)(P0 + r) * 128 + part * 8) =
            *(const v4u*)(hh + r * 136 + part * 8);
    }
}

// ---------------- fused r-projection + attention (flash-style) -------------
// grid (N/16, B), block 256. wave w owns head w. Online softmax over 12 j-tiles.
// rel_pe A-tiles staged in LDS (two 128-row chunks) so all 8 heads share one
// global read; v tile staged too. Next tile prefetched (global_prefetch_b8).
__global__ void __launch_bounds__(256) attn_kernel(
    const float* __restrict__ q, const float* __restrict__ k,
    const float* __restrict__ v, const _Float16* __restrict__ relpe,
    const _Float16* __restrict__ wrelt, const float* __restrict__ brel,
    const unsigned char* __restrict__ mask, float* __restrict__ ao) {
    __shared__ _Float16 as_t[128 * 136];          // 34 KB: 8-ii chunk of rel_pe
    __shared__ float    sc_t[HH][16][16];         //  8 KB: per-j-tile scores
    __shared__ float    v_s[16 * 128];            //  8 KB: v tile
    const int b = blockIdx.y, it = blockIdx.x;
    const int tid = threadIdx.x;
    const int w = tid >> 5, lane = tid & 31;
    const int l16 = lane & 15, g = lane >> 4;

    // preload wrel B-fragments for head w (cols w*16 .. w*16+15)
    v16h bf[4];
    {
        const _Float16* brow = wrelt + (size_t)(w * 16 + l16) * 128;
        #pragma unroll
        for (int kb = 0; kb < 4; ++kb) bf[kb] = load_frag(brow, kb);
    }
    const float brelv = brel[w * 16 + l16];
    // preload q[i, head w, d=l16] for the 16 i's of this tile
    float qv[16];
    #pragma unroll
    for (int ii = 0; ii < 16; ++ii)
        qv[ii] = q[((size_t)(b * NN + it * 16 + ii)) * DD + w * 16 + l16];

    // per-thread output state: 8 outputs each (o = tid + 256t -> ii=o>>7, c=o&127)
    float accum[8], mrun[8], lrun[8];
    #pragma unroll
    for (int t = 0; t < 8; ++t) { accum[t] = 0.f; lrun[t] = 0.f; mrun[t] = -1e30f; }

    for (int jt = 0; jt < 12; ++jt) {
        // stage v tile for this j-block (after prior barrier)
        for (int e = tid; e < 16 * 128; e += 256)
            v_s[e] = v[((size_t)(b * NN + jt * 16 + (e >> 7))) * DD + (e & 127)];
        // k[j = jt*16 + rr + 8g, head w, d = l16]
        float kreg[8];
        #pragma unroll
        for (int rr = 0; rr < 8; ++rr)
            kreg[rr] = k[((size_t)(b * NN + jt * 16 + rr + 8 * g)) * DD + w * 16 + l16];
        // prefetch next j-tile's rel_pe rows into cache
        if (jt + 1 < 12) {
            int pii = tid >> 4, pjj = tid & 15;
            const _Float16* p = relpe +
                ((size_t)(b * NN + it * 16 + pii) * NN + (jt + 1) * 16 + pjj) * 128;
            __builtin_prefetch(p, 0, 3);
            __builtin_prefetch(p + 64, 0, 3);
        }

        for (int half = 0; half < 2; ++half) {
            if (half) __syncthreads();            // half-0 readers done
            {   // stage 128 rel_pe rows (ii8 in [0,8), jj in [0,16)) into LDS
                int r = tid >> 1, seg = tid & 1;  // 32 halves (64 B) per thread
                const _Float16* src = relpe +
                    ((size_t)(b * NN + it * 16 + half * 8 + (r >> 4)) * NN
                     + jt * 16 + (r & 15)) * 128 + seg * 64;
                v4u* dst = (v4u*)(as_t + r * 136 + seg * 64);
                const v4u* s4 = (const v4u*)src;
                #pragma unroll
                for (int u = 0; u < 4; ++u) dst[u] = s4[u];
            }
            __syncthreads();
            for (int ii8 = 0; ii8 < 8; ++ii8) {
                const int ii = half * 8 + ii8;
                const _Float16* arow = as_t + (ii8 * 16 + l16) * 136;
                v8f acc = {};
                #pragma unroll
                for (int kb = 0; kb < 4; ++kb)
                    acc = wmma_f16(load_frag(arow, kb), bf[kb], acc);
                // scores(h=w, i, j) = sum_d (r + brel + k)*q ; reduce 16 d-lanes
                float val[8];
                #pragma unroll
                for (int rr = 0; rr < 8; ++rr) {
                    float t = (acc[rr] + brelv + kreg[rr]) * qv[ii];
                    #pragma unroll
                    for (int m = 8; m >= 1; m >>= 1) t += __shfl_xor(t, m, 32);
                    val[rr] = t;
                }
                if (l16 < 8) {
                    int jj = l16 + 8 * g;
                    float s = val[l16] * 0.25f;               // 1/sqrt(DH)
                    if (!mask[b * NN + jt * 16 + jj]) s += -1e9f;
                    sc_t[w][ii][jj] = s;
                }
            }
        }
        __syncthreads();
        // online-softmax update of per-thread accumulators
        #pragma unroll
        for (int t = 0; t < 8; ++t) {
            int o = tid + 256 * t;
            int ii = o >> 7, c = o & 127, h = c >> 4;
            float tm = -1e30f;
            #pragma unroll
            for (int jj = 0; jj < 16; ++jj) tm = fmaxf(tm, sc_t[h][ii][jj]);
            float mn = fmaxf(mrun[t], tm);
            float corr = __expf(mrun[t] - mn);
            float psum = 0.f, pv = 0.f;
            for (int jj = 0; jj < 16; ++jj) {
                float p = __expf(sc_t[h][ii][jj] - mn);
                psum += p;
                pv += p * v_s[jj * 128 + c];
            }
            lrun[t] = lrun[t] * corr + psum;
            accum[t] = accum[t] * corr + pv;
            mrun[t] = mn;
        }
        __syncthreads();
    }
    #pragma unroll
    for (int t = 0; t < 8; ++t) {
        int o = tid + 256 * t;
        int ii = o >> 7, c = o & 127;
        ao[((size_t)(b * NN + it * 16 + ii)) * DD + c] = accum[t] / lrun[t];
    }
}

// ---------------- host orchestration ----------------
extern "C" void kernel_launch(void* const* d_in, const int* in_sizes, int n_in,
                              void* d_out, int out_size, void* d_ws, size_t ws_size,
                              hipStream_t stream) {
    (void)in_sizes; (void)n_in; (void)out_size; (void)ws_size;
    const float* local_embed = (const float*)d_in[0];
    const float* pos         = (const float*)d_in[1];
    const float* heading     = (const float*)d_in[2];
    const unsigned char* msk = (const unsigned char*)d_in[3];
    const float* emb_w1 = (const float*)d_in[4];
    const float* emb_b1 = (const float*)d_in[5];
    const float* emb_g1 = (const float*)d_in[6];
    const float* emb_bb1= (const float*)d_in[7];
    const float* emb_w2 = (const float*)d_in[8];
    const float* emb_b2 = (const float*)d_in[9];
    const float* emb_g2 = (const float*)d_in[10];
    const float* emb_bb2= (const float*)d_in[11];
    const float* emb_w3 = (const float*)d_in[12];
    const float* emb_b3 = (const float*)d_in[13];
    const float* emb_g3 = (const float*)d_in[14];
    const float* emb_bb3= (const float*)d_in[15];
    const float* ln1_g = (const float*)d_in[16];
    const float* ln1_b = (const float*)d_in[17];
    const float* wq = (const float*)d_in[18];
    const float* bq = (const float*)d_in[19];
    const float* wk = (const float*)d_in[20];
    const float* bk = (const float*)d_in[21];
    const float* wv = (const float*)d_in[22];
    const float* bv = (const float*)d_in[23];
    const float* wrel = (const float*)d_in[24];
    const float* brel = (const float*)d_in[25];
    const float* wo = (const float*)d_in[26];
    const float* bo = (const float*)d_in[27];
    const float* ln2_g = (const float*)d_in[28];
    const float* ln2_b = (const float*)d_in[29];
    const float* fw1 = (const float*)d_in[30];
    const float* fb1 = (const float*)d_in[31];
    const float* fw2 = (const float*)d_in[32];
    const float* fb2 = (const float*)d_in[33];

    // workspace carve-up
    char* ws = (char*)d_ws;
    size_t off = 0;
    auto carve = [&](size_t bytes) -> char* {
        char* p = ws + off;
        off += (bytes + 255) & ~(size_t)255;
        return p;
    };
    _Float16* relpe = (_Float16*)carve((size_t)NPAIR * DD * 2);       // 75.5 MB
    _Float16* w2t   = (_Float16*)carve(128 * 128 * 2);
    _Float16* w3t   = (_Float16*)carve(128 * 128 * 2);
    _Float16* wq_t  = (_Float16*)carve((size_t)LL * 128 * 128 * 2);
    _Float16* wk_t  = (_Float16*)carve((size_t)LL * 128 * 128 * 2);
    _Float16* wv_t  = (_Float16*)carve((size_t)LL * 128 * 128 * 2);
    _Float16* wrel_t= (_Float16*)carve((size_t)LL * 128 * 128 * 2);
    _Float16* wo_t  = (_Float16*)carve((size_t)LL * 128 * 128 * 2);
    _Float16* fw1_t = (_Float16*)carve((size_t)LL * 128 * FFD * 2);
    _Float16* fw2_t = (_Float16*)carve((size_t)LL * FFD * 128 * 2);
    float* xbuf = (float*)carve((size_t)MM * DD * 4);
    float* xn   = (float*)carve((size_t)MM * DD * 4);
    float* qb   = (float*)carve((size_t)MM * DD * 4);
    float* kb2  = (float*)carve((size_t)MM * DD * 4);
    float* vb   = (float*)carve((size_t)MM * DD * 4);
    float* aob  = (float*)carve((size_t)MM * DD * 4);
    float* hff  = (float*)carve((size_t)MM * FFD * 4);

    auto tc = [&](const float* src, _Float16* dst, int R, int C) {
        int total = R * C;
        tc_kernel<<<(total + 255) / 256, 256, 0, stream>>>(src, dst, R, C);
    };
    tc(emb_w2, w2t, 128, 128);
    tc(emb_w3, w3t, 128, 128);
    for (int l = 0; l < LL; ++l) {
        tc(wq   + (size_t)l * 16384, wq_t   + (size_t)l * 16384, 128, 128);
        tc(wk   + (size_t)l * 16384, wk_t   + (size_t)l * 16384, 128, 128);
        tc(wv   + (size_t)l * 16384, wv_t   + (size_t)l * 16384, 128, 128);
        tc(wrel + (size_t)l * 16384, wrel_t + (size_t)l * 16384, 128, 128);
        tc(wo   + (size_t)l * 16384, wo_t   + (size_t)l * 16384, 128, 128);
        tc(fw1  + (size_t)l * 65536, fw1_t  + (size_t)l * 65536, 128, FFD);
        tc(fw2  + (size_t)l * 65536, fw2_t  + (size_t)l * 65536, FFD, 128);
    }

    relpe_kernel<<<NPAIR / 16, 256, 0, stream>>>(
        pos, heading, emb_w1, emb_b1, emb_g1, emb_bb1,
        w2t, emb_b2, emb_g2, emb_bb2, w3t, emb_b3, emb_g3, emb_bb3, relpe);

    hipMemcpyAsync(xbuf, local_embed, (size_t)MM * DD * 4,
                   hipMemcpyDeviceToDevice, stream);

    for (int l = 0; l < LL; ++l) {
        ln_kernel<<<MM / 8, 256, 0, stream>>>(xbuf, ln1_g + l * DD, ln1_b + l * DD, xn, MM);
        gemm16_kernel<<<dim3(MM / 16, 1), 256, 0, stream>>>(
            xn, wq_t + (size_t)l * 16384, bq + l * DD, nullptr, qb, 128, 128, 0);
        gemm16_kernel<<<dim3(MM / 16, 1), 256, 0, stream>>>(
            xn, wk_t + (size_t)l * 16384, bk + l * DD, nullptr, kb2, 128, 128, 0);
        gemm16_kernel<<<dim3(MM / 16, 1), 256, 0, stream>>>(
            xn, wv_t + (size_t)l * 16384, bv + l * DD, nullptr, vb, 128, 128, 0);
        attn_kernel<<<dim3(NN / 16, BB), 256, 0, stream>>>(
            qb, kb2, vb, relpe, wrel_t + (size_t)l * 16384, brel + l * DD, msk, aob);
        gemm16_kernel<<<dim3(MM / 16, 1), 256, 0, stream>>>(
            aob, wo_t + (size_t)l * 16384, bo + l * DD, xbuf, xbuf, 128, 128, 0);
        ln_kernel<<<MM / 8, 256, 0, stream>>>(xbuf, ln2_g + l * DD, ln2_b + l * DD, xn, MM);
        gemm16_kernel<<<dim3(MM / 16, FFD / 128), 256, 0, stream>>>(
            xn, fw1_t + (size_t)l * 65536, fb1 + l * FFD, nullptr, hff, 128, FFD, 1);
        gemm16_kernel<<<dim3(MM / 16, 1), 256, 0, stream>>>(
            hff, fw2_t + (size_t)l * 65536, fb2 + l * DD, xbuf, xbuf, FFD, 128, 0);
    }

    hipMemcpyAsync(d_out, xbuf, (size_t)MM * DD * 4,
                   hipMemcpyDeviceToDevice, stream);
}
